// ShowAndTell_53558242181142
// MI455X (gfx1250) — matrix operand order repaired and checked by reading, not compiled, not run on hardware
//
#include <hip/hip_runtime.h>
#include <hip/hip_bf16.h>

// ---------------------------------------------------------------------------
// Types for CDNA5 WMMA (wave32, v_wmma_f32_16x16x32_bf16) + TDM descriptors
// ---------------------------------------------------------------------------
typedef __bf16 bf16;
typedef __attribute__((ext_vector_type(16))) bf16         v16bf;
typedef __attribute__((ext_vector_type(8)))  float        v8f;
typedef __attribute__((ext_vector_type(4)))  unsigned int u32x4;
typedef __attribute__((ext_vector_type(8)))  int          i32x8;
typedef __attribute__((ext_vector_type(4)))  int          i32x4;

union Frag2x16B { v16bf v; uint4 q[2]; };

// ---------------------------------------------------------------------------
// TDM: stage a 2D bf16 panel (rows x elems, row stride in elements) from
// global memory into LDS at byte offset lds_off. D# layout per ISA 8.3/8.4.
// Issue from ONE wave only (TDM ignores EXEC), then s_wait_tensorcnt.
// This toolchain exposes the 6-arg builtin: (g0, g1, g2, g3, g_ext, cpol).
// ---------------------------------------------------------------------------
__device__ __forceinline__ void tdm_load_panel_to_lds(const void* gptr, unsigned lds_off,
                                                      unsigned elems, unsigned rows,
                                                      unsigned row_stride_elems) {
  unsigned long long ga = (unsigned long long)(size_t)gptr;
  u32x4 g0;
  g0[0] = 1u;                                              // count=1, user mode
  g0[1] = lds_off;                                         // lds_addr (bytes)
  g0[2] = (unsigned)(ga & 0xFFFFFFFFu);                    // global_addr[31:0]
  g0[3] = (unsigned)((ga >> 32) & 0x01FFFFFFu)             // global_addr[56:32]
        | (2u << 30);                                      // type=2 ("image")
  i32x8 g1;
  g1[0] = (int)(1u << 16);                                 // wg_mask=0, data_size=1 (2B)
  g1[1] = (int)((elems & 0xFFFFu) << 16);                  // tensor_dim0[15:0]
  g1[2] = (int)(((elems >> 16) & 0xFFFFu)                  // tensor_dim0[31:16]
        | ((rows & 0xFFFFu) << 16));                       // tensor_dim1[15:0]
  g1[3] = (int)(((rows >> 16) & 0xFFFFu)                   // tensor_dim1[31:16]
        | ((elems & 0xFFFFu) << 16));                      // tile_dim0 = elems
  g1[4] = (int)(rows & 0xFFFFu);                           // tile_dim1 = rows, tile_dim2=0
  g1[5] = (int)row_stride_elems;                           // tensor_dim0_stride[31:0]
  g1[6] = 0;                                               // stride hi, dim1_stride lo
  g1[7] = 0;
  i32x4 z4 = {0, 0, 0, 0};                                 // groups 2/3 unused (2D)
  i32x8 z8 = {0, 0, 0, 0, 0, 0, 0, 0};                     // extended group (unused)
  __builtin_amdgcn_tensor_load_to_lds(g0, g1, z4, z4, z8, 0);
}

// ---------------------------------------------------------------------------
// Fragment loaders
// ---------------------------------------------------------------------------
// A-frag (16x32 bf16) from an LDS panel laid out [rows][K] row-major.
// ISA layout: lane -> M = lane&15; K halves {0..7,16..23} (+8 for hi lanes).
__device__ __forceinline__ v16bf lds_a_frag(const bf16* ldsA, int Kld,
                                            int m_local, int k0, int lane) {
  const int lhi = (lane >> 4) & 1;
  const bf16* row = ldsA + (size_t)m_local * Kld + k0 + lhi * 8;
  Frag2x16B f;
  f.q[0] = *(const uint4*)(row);        // K halves +0..+7
  f.q[1] = *(const uint4*)(row + 16);   // K halves +16..+23
  return f.v;
}

// B-frag (32x16 bf16), K-major in global memory: lane holds column N = lane&15,
// K halves {0..7,16..23} (+8 for hi lanes) -> 16 strided loads.
__device__ __forceinline__ v16bf load_b_frag(const bf16* __restrict__ B, int ldb,
                                             int k0, int n0, int lane) {
  const int lhi = (lane >> 4) & 1;
  const bf16* col = B + (size_t)(k0 + lhi * 8) * ldb + n0 + (lane & 15);
  v16bf v;
#pragma unroll
  for (int i = 0; i < 8; ++i) {
    v[i] = col[(size_t)i * ldb];
    v[8 + i] = col[(size_t)(16 + i) * ldb];
  }
  return v;
}

__device__ __forceinline__ v8f wmma_bf16(v16bf a, v16bf b, v8f c) {
  return __builtin_amdgcn_wmma_f32_16x16x32_bf16(false, a, false, b,
                                                 (short)0, c, false, false);
}

// ---------------------------------------------------------------------------
// Pack / convert kernels (one-time, bandwidth-trivial vs 508MB output)
// ---------------------------------------------------------------------------
__global__ void cvt_wihT_kernel(const float* __restrict__ W_ih, bf16* __restrict__ dst) {
  int idx = blockIdx.x * blockDim.x + threadIdx.x;           // 320*1536
  if (idx >= 320 * 1536) return;
  int k = idx / 1536, n = idx - k * 1536;                    // dst[k][n] = W_ih[n][k]
  float v = (k < 300) ? W_ih[(size_t)n * 300 + k] : 0.f;
  dst[idx] = (bf16)v;
}

__global__ void cvt_whhT_kernel(const float* __restrict__ W_hh, bf16* __restrict__ dst) {
  int idx = blockIdx.x * blockDim.x + threadIdx.x;           // 512*1536
  if (idx >= 512 * 1536) return;
  int k = idx / 1536, j = idx - k * 1536;                    // dst[k][j] = W_hh[j][k]
  dst[idx] = (bf16)W_hh[(size_t)j * 512 + k];
}

__global__ void cvt_wcls_kernel(const float* __restrict__ W_cls, bf16* __restrict__ dst, int total) {
  int idx = blockIdx.x * blockDim.x + threadIdx.x;           // 512*32000, already K-major
  if (idx >= total) return;
  dst[idx] = (bf16)W_cls[idx];
}

__global__ void pack_x_kernel(const float* __restrict__ lang, bf16* __restrict__ dst) {
  int idx = blockIdx.x * blockDim.x + threadIdx.x;           // 3968*320 (K padded 300->320)
  if (idx >= 3968 * 320) return;
  int m = idx / 320, k = idx - m * 320;
  int t = m >> 7, b = m & 127;                               // m = t*128 + b
  float v = (k < 300) ? lang[((size_t)b * 32 + t) * 300 + k] : 0.f;
  dst[idx] = (bf16)v;
}

// hidden0 = relu([g|t] @ W_feat + b_feat)   (tiny: 33 MFLOP, fp32 VALU)
__global__ void hidden0_kernel(const float* __restrict__ g, const float* __restrict__ t,
                               const float* __restrict__ Wf, const float* __restrict__ bf_,
                               float* __restrict__ H32_0, bf16* __restrict__ H0B) {
  int idx = blockIdx.x * blockDim.x + threadIdx.x;           // 128*512
  if (idx >= 128 * 512) return;
  int b = idx >> 9, j = idx & 511;
  const float* gb = g + (size_t)b * 128;
  const float* tb = t + (size_t)b * 128;
  float acc = bf_[j];
#pragma unroll 4
  for (int k = 0; k < 128; ++k) acc = fmaf(gb[k], Wf[(size_t)k * 512 + j], acc);
#pragma unroll 4
  for (int k = 0; k < 128; ++k) acc = fmaf(tb[k], Wf[(size_t)(128 + k) * 512 + j], acc);
  acc = fmaxf(acc, 0.f);
  H32_0[idx] = acc;
  H0B[idx] = (bf16)acc;
}

// ---------------------------------------------------------------------------
// bf16 WMMA GEMM:  C[M,N](f32) = A[M,K] * B[K,N] + bias[N]
// Block = 256 threads = 8 waves; block tile 32(M) x 256(N); wave tile 16x64.
// A panel (32 x K bf16, <=32KB) staged to LDS once per block via TDM.
// REMAP=true scatters rows m = t*128+b into out[b][t][:] (classifier).
// ---------------------------------------------------------------------------
template <bool REMAP, int K>
__global__ __launch_bounds__(256) void gemm_bf16_kernel(
    const bf16* __restrict__ A,            // [M,K] row-major (lda = K)
    const bf16* __restrict__ B, int ldb,   // [K,N] K-major
    const float* __restrict__ bias,
    float* __restrict__ C, int ldc) {
  __shared__ __align__(16) bf16 ldsA[32 * K];

  const int lane = threadIdx.x & 31;
  const int wave = threadIdx.x >> 5;
  const int M0 = blockIdx.y * 32;
  const int m0 = M0 + (wave & 1) * 16;
  const int n0 = blockIdx.x * 256 + (wave >> 1) * 64;
  const int m_local = (wave & 1) * 16 + (lane & 15);

  // ---- TDM: one wave DMAs the 32 x K A-panel into LDS (EXEC ignored by TDM,
  //      so gate with a whole-wave branch), then waits TENSORcnt. ----
  if (threadIdx.x < 32) {
    tdm_load_panel_to_lds(A + (size_t)M0 * K, /*lds_off=*/0u, K, 32, K);
    __builtin_amdgcn_s_wait_tensorcnt(0);
  }
  __syncthreads();

  const v8f vzero = {0.f, 0.f, 0.f, 0.f, 0.f, 0.f, 0.f, 0.f};
  v8f acc0 = vzero, acc1 = vzero, acc2 = vzero, acc3 = vzero;

#pragma unroll
  for (int k0 = 0; k0 < K; k0 += 32) {
    if (k0 + 32 < K) {  // gfx1250 global_prefetch_b8: pull next B K-slab forward
      __builtin_prefetch(B + (size_t)(k0 + 32 + lane) * ldb + n0, 0, 1);
    }
    // Independent registers for all four B fragments -> loads issue together.
    v16bf b0 = load_b_frag(B, ldb, k0, n0 + 0,  lane);
    v16bf b1 = load_b_frag(B, ldb, k0, n0 + 16, lane);
    v16bf b2 = load_b_frag(B, ldb, k0, n0 + 32, lane);
    v16bf b3 = load_b_frag(B, ldb, k0, n0 + 48, lane);
    v16bf a  = lds_a_frag(ldsA, K, m_local, k0, lane);
    acc0 = wmma_bf16(a, b0, acc0);
    acc1 = wmma_bf16(a, b1, acc1);
    acc2 = wmma_bf16(a, b2, acc2);
    acc3 = wmma_bf16(a, b3, acc3);
  }

  const int lhi8 = ((lane >> 4) & 1) * 8;
  v8f accs[4] = {acc0, acc1, acc2, acc3};
#pragma unroll
  for (int s = 0; s < 4; ++s) {
    const int col = n0 + s * 16 + (lane & 15);
    const float bv = bias[col];
#pragma unroll
    for (int r = 0; r < 8; ++r) {
      const int m = m0 + lhi8 + r;
      const float v = accs[s][r] + bv;
      if (REMAP) {
        const int b_ = m & 127, t_ = m >> 7;           // m = t*128 + b
        C[((size_t)b_ * 31 + t_) * 32000 + col] = v;   // out[b][t][v]
      } else {
        C[(size_t)m * ldc + col] = v;
      }
    }
  }
}

// ---------------------------------------------------------------------------
// Fused GRU step: gh = h @ W_hh^T (WMMA, K=512), gates in fp32, h_{t+1} out.
// 1 wave/block; grid (H/64=8, B/16=8). h-panel (16x512 bf16) staged via TDM.
// ---------------------------------------------------------------------------
__global__ __launch_bounds__(32) void gru_step_kernel(
    const bf16*  __restrict__ Hb_prev,   // [128,512] bf16 (A operand)
    const float* __restrict__ H32_prev,  // [128,512] f32
    const float* __restrict__ GI_t,      // [128,1536] f32, includes b_ih
    const bf16*  __restrict__ WhhT,      // [512,1536] bf16 (K-major)
    const float* __restrict__ b_hh,      // [1536]
    float* __restrict__ H32_next,        // [128,512]
    bf16*  __restrict__ Hb_next) {       // [128,512] bf16 (= HB slot t)
  __shared__ __align__(16) bf16 ldsH[16 * 512];

  const int lane = threadIdx.x & 31;
  const int m0 = blockIdx.y * 16;        // batch rows
  const int n0 = blockIdx.x * 64;        // hidden cols (within H=512)

  // Single wave per block: DMA h-panel, wait, no barrier needed.
  tdm_load_panel_to_lds(Hb_prev + (size_t)m0 * 512, 0u, 512, 16, 512);
  __builtin_amdgcn_s_wait_tensorcnt(0);

  const v8f vzero = {0.f, 0.f, 0.f, 0.f, 0.f, 0.f, 0.f, 0.f};
  v8f acc[3][4];
#pragma unroll
  for (int g = 0; g < 3; ++g)
#pragma unroll
    for (int s = 0; s < 4; ++s) acc[g][s] = vzero;

#pragma unroll
  for (int k0 = 0; k0 < 512; k0 += 32) {
    v16bf a = lds_a_frag(ldsH, 512, lane & 15, k0, lane);
#pragma unroll
    for (int g = 0; g < 3; ++g) {
      v16bf b0 = load_b_frag(WhhT, 1536, k0, g * 512 + n0 + 0,  lane);
      v16bf b1 = load_b_frag(WhhT, 1536, k0, g * 512 + n0 + 16, lane);
      v16bf b2 = load_b_frag(WhhT, 1536, k0, g * 512 + n0 + 32, lane);
      v16bf b3 = load_b_frag(WhhT, 1536, k0, g * 512 + n0 + 48, lane);
      acc[g][0] = wmma_bf16(a, b0, acc[g][0]);
      acc[g][1] = wmma_bf16(a, b1, acc[g][1]);
      acc[g][2] = wmma_bf16(a, b2, acc[g][2]);
      acc[g][3] = wmma_bf16(a, b3, acc[g][3]);
    }
  }

  const int lhi8 = ((lane >> 4) & 1) * 8;
#pragma unroll
  for (int s = 0; s < 4; ++s) {
    const int j = n0 + s * 16 + (lane & 15);
#pragma unroll
    for (int r = 0; r < 8; ++r) {
      const int m = m0 + lhi8 + r;
      const float* gi = GI_t + (size_t)m * 1536;
      const float ghr = acc[0][s][r] + b_hh[j];
      const float ghz = acc[1][s][r] + b_hh[512 + j];
      const float ghn = acc[2][s][r] + b_hh[1024 + j];
      const float rr = 1.f / (1.f + __expf(-(gi[j] + ghr)));
      const float zz = 1.f / (1.f + __expf(-(gi[512 + j] + ghz)));
      const float nn = tanhf(gi[1024 + j] + rr * ghn);
      const float ho = H32_prev[(size_t)m * 512 + j];
      const float hn = (1.f - zz) * nn + zz * ho;
      H32_next[(size_t)m * 512 + j] = hn;
      Hb_next[(size_t)m * 512 + j] = (bf16)hn;
    }
  }
}

// ---------------------------------------------------------------------------
// Launch
// ---------------------------------------------------------------------------
extern "C" void kernel_launch(void* const* d_in, const int* in_sizes, int n_in,
                              void* d_out, int out_size, void* d_ws, size_t ws_size,
                              hipStream_t stream) {
  const float* g_feat = (const float*)d_in[0];   // [128,128]
  const float* t_feat = (const float*)d_in[1];   // [128,128]
  const float* lang   = (const float*)d_in[2];   // [128,32,300]
  // d_in[3] = lang_len (all == T, unused)
  const float* W_feat = (const float*)d_in[4];   // [256,512]
  const float* b_feat = (const float*)d_in[5];   // [512]
  const float* W_ih   = (const float*)d_in[6];   // [1536,300]
  const float* W_hh   = (const float*)d_in[7];   // [1536,512]
  const float* b_ih   = (const float*)d_in[8];   // [1536]
  const float* b_hh   = (const float*)d_in[9];   // [1536]
  const float* W_cls  = (const float*)d_in[10];  // [512,32000]
  const float* b_cls  = (const float*)d_in[11];  // [32000]
  float* out = (float*)d_out;                    // [128,31,32000]

  // ---- workspace carve-out (256B aligned), ~67 MB total ----
  char* ws = (char*)d_ws;
  size_t off = 0;
  auto carve = [&](size_t bytes) -> void* {
    void* p = ws + off;
    off += (bytes + 255) & ~(size_t)255;
    return p;
  };
  bf16*  XB    = (bf16*)carve((size_t)3968 * 320 * 2);    // packed x, K padded to 320
  bf16*  WihT  = (bf16*)carve((size_t)320 * 1536 * 2);    // W_ih^T  (K-major)
  bf16*  WhhT  = (bf16*)carve((size_t)512 * 1536 * 2);    // W_hh^T  (K-major)
  bf16*  WclsB = (bf16*)carve((size_t)512 * 32000 * 2);   // W_cls (already K-major)
  float* GI    = (float*)carve((size_t)3968 * 1536 * 4);  // precomputed input gates
  bf16*  HB    = (bf16*)carve((size_t)3968 * 512 * 2);    // h_t (bf16) for t=1..31
  bf16*  H0B   = (bf16*)carve((size_t)128 * 512 * 2);     // h_0 bf16
  float* H32a  = (float*)carve((size_t)128 * 512 * 4);    // fp32 h ping
  float* H32b  = (float*)carve((size_t)128 * 512 * 4);    // fp32 h pong
  (void)ws_size; (void)in_sizes; (void)n_in; (void)out_size;

  // ---- 1) convert / transpose / pack ----
  cvt_wihT_kernel<<<(320 * 1536 + 255) / 256, 256, 0, stream>>>(W_ih, WihT);
  cvt_whhT_kernel<<<(512 * 1536 + 255) / 256, 256, 0, stream>>>(W_hh, WhhT);
  cvt_wcls_kernel<<<(512 * 32000 + 255) / 256, 256, 0, stream>>>(W_cls, WclsB, 512 * 32000);
  pack_x_kernel<<<(3968 * 320 + 255) / 256, 256, 0, stream>>>(lang, XB);

  // ---- 2) hidden0 = relu([g|t] @ W_feat + b_feat) ----
  hidden0_kernel<<<(128 * 512 + 255) / 256, 256, 0, stream>>>(
      g_feat, t_feat, W_feat, b_feat, H32a, H0B);

  // ---- 3) gi = x @ W_ih^T + b_ih, batched over all 31 steps (M=3968,N=1536,K=320) ----
  gemm_bf16_kernel<false, 320><<<dim3(1536 / 256, 3968 / 32), 256, 0, stream>>>(
      XB, WihT, 1536, b_ih, GI, 1536);

  // ---- 4) 31 sequential fused GRU steps (stream order = recurrence dependency) ----
  for (int t = 0; t < 31; ++t) {
    const bf16*  hb_prev  = (t == 0) ? H0B : (HB + (size_t)(t - 1) * 128 * 512);
    const float* h32_prev = (t & 1) ? H32b : H32a;
    float*       h32_next = (t & 1) ? H32a : H32b;
    gru_step_kernel<<<dim3(512 / 64, 128 / 16), 32, 0, stream>>>(
        hb_prev, h32_prev, GI + (size_t)t * 128 * 1536, WhhT, b_hh,
        h32_next, HB + (size_t)t * 128 * 512);
  }

  // ---- 5) logits = H @ W_cls + b_cls (M=3968,N=32000,K=512), scatter to [B,T-1,V] ----
  gemm_bf16_kernel<true, 512><<<dim3(32000 / 256, 3968 / 32), 256, 0, stream>>>(
      HB, WclsB, 32000, b_cls, out, 32000);
}
